// SLRModel_83030307766724
// MI455X (gfx1250) — matrix operand order repaired and checked by reading, not compiled, not run on hardware
//
#include <hip/hip_runtime.h>
#include <hip/hip_bf16.h>

#define T_DIM 8192
#define C_DIM 1024
#define L_SEG 128
#define NB_B  3
#define EPSF  1e-6f

typedef __attribute__((ext_vector_type(16))) __bf16 v16bf;
typedef __attribute__((ext_vector_type(8)))  __bf16 v8bf;
typedef __attribute__((ext_vector_type(4)))  __bf16 v4bf;
typedef __attribute__((ext_vector_type(8)))  float  v8f;

__device__ __forceinline__ v16bf cat8(v8bf lo, v8bf hi) {
    v16bf r;
#pragma unroll
    for (int i = 0; i < 8; ++i) { r[i] = lo[i]; r[8 + i] = hi[i]; }
    return r;
}

// ---------------------------------------------------------------------------
// Tiled bf16 WMMA GEMM:  Out[m,n] = sum_k A[m,k] * W[n,k]  (+ bias[n])
//   A: [T_DIM x C_DIM] row-major (float or bf16), per-batch stride a_bstride
//   W: [C_DIM x C_DIM] row-major float weights, per-batch stride b_bstride
//   Out: bf16 (OUT_BF16) or float(+bias)
// Block: 256 thr (8 waves). Tile 128x128, KT=32. Wave = 2x4 subtiles 16x16.
// ---------------------------------------------------------------------------
template <typename TIn, bool OUT_BF16>
__global__ void wmma_gemm(const TIn* __restrict__ A, long long a_bstride,
                          const float* __restrict__ W, long long b_bstride,
                          const float* __restrict__ bias,
                          void* __restrict__ OutV, long long o_bstride) {
    constexpr int MT = 128, NT = 128, KT = 32, KP = KT + 8;
    __shared__ __bf16 As[MT][KP];
    __shared__ __bf16 Bs[NT][KP];

    const int b  = blockIdx.z;
    const int m0 = blockIdx.x * MT;
    const int n0 = blockIdx.y * NT;
    const TIn*   Ab = A + (size_t)b * a_bstride;
    const float* Wb = W + (size_t)b * b_bstride;

    const int tid  = threadIdx.x;
    const int lane = tid & 31;
    const int wave = tid >> 5;
    const int mblk = (wave & 3) * 32;   // 2 M-subtiles
    const int nblk = (wave >> 2) * 64;  // 4 N-subtiles
    const int l16  = lane & 15;
    const int half = (lane >> 4);       // 0 or 1

    v8f acc[2][4] = {};

    for (int k0 = 0; k0 < C_DIM; k0 += KT) {
        if (k0) __syncthreads();
        // ---- stage A tile (convert to bf16 if needed) ----
        if constexpr (__is_same(TIn, float)) {
#pragma unroll
            for (int u = tid; u < MT * KT / 4; u += 256) {
                const int row = u >> 3, c4 = (u & 7) * 4;
                const float4 f = *(const float4*)(Ab + (size_t)(m0 + row) * C_DIM + k0 + c4);
                v4bf h; h[0] = (__bf16)f.x; h[1] = (__bf16)f.y; h[2] = (__bf16)f.z; h[3] = (__bf16)f.w;
                *(v4bf*)&As[row][c4] = h;
            }
        } else {
#pragma unroll
            for (int u = tid; u < MT * KT / 8; u += 256) {
                const int row = u >> 2, c8 = (u & 3) * 8;
                *(v8bf*)&As[row][c8] =
                    *(const v8bf*)(Ab + (size_t)(m0 + row) * C_DIM + k0 + c8);
            }
        }
        // ---- stage B tile: rows of W (B[k,n] = W[n,k] -> W rows are contiguous) ----
#pragma unroll
        for (int u = tid; u < NT * KT / 4; u += 256) {
            const int row = u >> 3, c4 = (u & 7) * 4;
            const float4 f = *(const float4*)(Wb + (size_t)(n0 + row) * C_DIM + k0 + c4);
            v4bf h; h[0] = (__bf16)f.x; h[1] = (__bf16)f.y; h[2] = (__bf16)f.z; h[3] = (__bf16)f.w;
            *(v4bf*)&Bs[row][c4] = h;
        }
        __syncthreads();

        // ---- A fragments: lanes 0-15 hold K{0-7,16-23}, lanes 16-31 K{8-15,24-31} ----
        v16bf afrag[2], bfrag[4];
        const int ha = half * 8;
#pragma unroll
        for (int mi = 0; mi < 2; ++mi) {
            const int row = mblk + mi * 16 + l16;
            afrag[mi] = cat8(*(const v8bf*)&As[row][ha],
                             *(const v8bf*)&As[row][16 + ha]);
        }
        // ---- B fragments: lanes 0-15 hold K{0-7,16-23}, lanes 16-31 K{8-15,24-31} ----
        const int hb = half * 8;
#pragma unroll
        for (int ni = 0; ni < 4; ++ni) {
            const int col = nblk + ni * 16 + l16;
            bfrag[ni] = cat8(*(const v8bf*)&Bs[col][hb],
                             *(const v8bf*)&Bs[col][hb + 16]);
        }
#pragma unroll
        for (int mi = 0; mi < 2; ++mi)
#pragma unroll
            for (int ni = 0; ni < 4; ++ni)
                acc[mi][ni] = __builtin_amdgcn_wmma_f32_16x16x32_bf16(
                    false, afrag[mi], false, bfrag[ni],
                    (short)0, acc[mi][ni], false, false);
    }

    // ---- epilogue: C/D layout lanes0-15 N=lane,M=v ; lanes16-31 N=lane-16,M=8+v ----
    const int moff = half * 8;
#pragma unroll
    for (int mi = 0; mi < 2; ++mi) {
#pragma unroll
        for (int ni = 0; ni < 4; ++ni) {
            const int n = n0 + nblk + ni * 16 + l16;
#pragma unroll
            for (int v = 0; v < 8; ++v) {
                const int m = m0 + mblk + mi * 16 + moff + v;
                const float val = acc[mi][ni][v];
                if constexpr (OUT_BF16) {
                    __bf16* O = (__bf16*)OutV + (size_t)b * o_bstride;
                    O[(size_t)m * C_DIM + n] = (__bf16)val;
                } else {
                    float* O = (float*)OutV + (size_t)b * o_bstride;
                    O[(size_t)m * C_DIM + n] = val + bias[(size_t)b * C_DIM + n];
                }
            }
        }
    }
}

// ---------------------------------------------------------------------------
// Cosine similarity of y[t] with y[t-1 mod T]. One block per (t, b).
// ---------------------------------------------------------------------------
__global__ void sim_kernel(const float* __restrict__ y, float* __restrict__ sim) {
    const int t = blockIdx.x, b = blockIdx.y, tid = threadIdx.x;
    const int tp = (t + T_DIM - 1) % T_DIM;
    const float* yc = y + ((size_t)b * T_DIM + t) * C_DIM;
    const float* yp = y + ((size_t)b * T_DIM + tp) * C_DIM;
    float d = 0.f, a2 = 0.f, b2 = 0.f;
    for (int c = tid; c < C_DIM; c += 256) {
        const float u = yc[c], w = yp[c];
        d += u * w; a2 += u * u; b2 += w * w;
    }
    __shared__ float sd[256], sa[256], sb[256];
    sd[tid] = d; sa[tid] = a2; sb[tid] = b2;
    __syncthreads();
    for (int s = 128; s > 0; s >>= 1) {
        if (tid < s) { sd[tid] += sd[tid + s]; sa[tid] += sa[tid + s]; sb[tid] += sb[tid + s]; }
        __syncthreads();
    }
    if (tid == 0) {
        float n1 = sqrtf(sa[0]); if (n1 < EPSF) n1 = EPSF;
        float n2 = sqrtf(sb[0]); if (n2 < EPSF) n2 = EPSF;
        sim[(size_t)b * T_DIM + t] = sd[0] / (n1 * n2);
    }
}

// ---------------------------------------------------------------------------
// Top-128 (tie -> lowest index, like lax.top_k), sort asc, emit starts/ends.
// One block per batch.
// ---------------------------------------------------------------------------
__global__ void topk_kernel(const float* __restrict__ sim,
                            int* __restrict__ starts, int* __restrict__ ends) {
    const int b = blockIdx.x, tid = threadIdx.x;
    __shared__ float sv[T_DIM];
    __shared__ float rbv[256];
    __shared__ int   rbi[256];
    __shared__ int   chosen[L_SEG];
    for (int i = tid; i < T_DIM; i += 256) sv[i] = sim[(size_t)b * T_DIM + i];
    __syncthreads();
    for (int it = 0; it < L_SEG; ++it) {
        float bv = -3.0e38f; int bi = T_DIM;
        for (int i = tid; i < T_DIM; i += 256) {
            const float v = sv[i];
            if (v > bv || (v == bv && i < bi)) { bv = v; bi = i; }
        }
        rbv[tid] = bv; rbi[tid] = bi;
        __syncthreads();
        for (int s = 128; s > 0; s >>= 1) {
            if (tid < s) {
                if (rbv[tid + s] > rbv[tid] ||
                    (rbv[tid + s] == rbv[tid] && rbi[tid + s] < rbi[tid])) {
                    rbv[tid] = rbv[tid + s]; rbi[tid] = rbi[tid + s];
                }
            }
            __syncthreads();
        }
        if (tid == 0) { chosen[it] = rbi[0]; sv[rbi[0]] = -3.0e38f; }
        __syncthreads();
    }
    if (tid == 0) {
        for (int i = 1; i < L_SEG; ++i) {          // insertion sort, ascending
            const int key = chosen[i];
            int j = i - 1;
            while (j >= 0 && chosen[j] > key) { chosen[j + 1] = chosen[j]; --j; }
            chosen[j + 1] = key;
        }
        int* st = starts + (size_t)b * L_SEG;
        int* en = ends   + (size_t)b * L_SEG;
        st[0] = 0;
        for (int i = 1; i <= L_SEG - 2; ++i) st[i] = chosen[i - 1];
        st[L_SEG - 1] = chosen[L_SEG - 1];
        for (int i = 0; i <= L_SEG - 2; ++i) en[i] = chosen[i];
        en[L_SEG - 1] = T_DIM - 1;
    }
}

// ---------------------------------------------------------------------------
// Segment means over y rows [start, end); len==0 -> y[start]. Block per (seg,b).
// ---------------------------------------------------------------------------
__global__ void seg_mean(const float* __restrict__ y,
                         const int* __restrict__ starts, const int* __restrict__ ends,
                         float* __restrict__ out) {
    const int seg = blockIdx.x, b = blockIdx.y, tid = threadIdx.x;
    const int s = starts[(size_t)b * L_SEG + seg];
    const int e = ends[(size_t)b * L_SEG + seg];
    const int len = e - s;
    const float* yb = y + (size_t)b * T_DIM * C_DIM;
    float* ob = out + ((size_t)b * L_SEG + seg) * C_DIM;
    if (len <= 0) {
        for (int c = tid; c < C_DIM; c += 256) ob[c] = yb[(size_t)s * C_DIM + c];
        return;
    }
    const float inv = 1.0f / (float)len;
    for (int c = tid; c < C_DIM; c += 256) {
        float acc = 0.f;
        for (int r = s; r < e; ++r) acc += yb[(size_t)r * C_DIM + c];
        ob[c] = acc * inv;
    }
}

extern "C" void kernel_launch(void* const* d_in, const int* in_sizes, int n_in,
                              void* d_out, int out_size, void* d_ws, size_t ws_size,
                              hipStream_t stream) {
    (void)in_sizes; (void)n_in; (void)out_size; (void)ws_size;
    const float* visual = (const float*)d_in[0];  // (NB, T, 1, C)
    const float* qkv_w  = (const float*)d_in[1];  // (NB, 3C, C)
    const float* proj_w = (const float*)d_in[2];  // (NB, C, C)
    const float* proj_b = (const float*)d_in[3];  // (NB, C)

    // workspace layout
    char* ws = (char*)d_ws;
    const size_t sz_v = (size_t)NB_B * T_DIM * C_DIM * sizeof(__bf16);
    const size_t sz_y = (size_t)NB_B * T_DIM * C_DIM * sizeof(float);
    const size_t sz_s = (size_t)NB_B * T_DIM * sizeof(float);
    __bf16* vbuf   = (__bf16*)ws;
    float*  ybuf   = (float*)(ws + sz_v);
    float*  simbuf = (float*)(ws + sz_v + sz_y);
    int*    starts = (int*)(ws + sz_v + sz_y + sz_s);
    int*    endsp  = starts + (size_t)NB_B * L_SEG;

    const dim3 gG(T_DIM / 128, C_DIM / 128, NB_B);
    // GEMM 1: v = x @ Wv^T  (Wv = qkv_w rows [2C, 3C))
    wmma_gemm<float, true><<<gG, 256, 0, stream>>>(
        visual, (long long)T_DIM * C_DIM,
        qkv_w + (size_t)2 * C_DIM * C_DIM, (long long)3 * C_DIM * C_DIM,
        nullptr, (void*)vbuf, (long long)T_DIM * C_DIM);
    // GEMM 2: y = v @ Wp^T + bp
    wmma_gemm<__bf16, false><<<gG, 256, 0, stream>>>(
        vbuf, (long long)T_DIM * C_DIM,
        proj_w, (long long)C_DIM * C_DIM,
        proj_b, (void*)ybuf, (long long)T_DIM * C_DIM);

    sim_kernel<<<dim3(T_DIM, NB_B), 256, 0, stream>>>(ybuf, simbuf);
    topk_kernel<<<NB_B, 256, 0, stream>>>(simbuf, starts, endsp);
    seg_mean<<<dim3(L_SEG, NB_B), 256, 0, stream>>>(ybuf, starts, endsp, (float*)d_out);
}
